// CNNxLSTMClassifier_76982993814109
// MI455X (gfx1250) — matrix operand order repaired and checked by reading, not compiled, not run on hardware
//
#include <hip/hip_runtime.h>
#include <cstdint>
#include <cstddef>

typedef unsigned short u16;
typedef __bf16 bf16_t;
typedef bf16_t v16bf __attribute__((ext_vector_type(16)));
typedef u16    v16u  __attribute__((ext_vector_type(16)));
typedef float  v8f   __attribute__((ext_vector_type(8)));

#define EPSLN 1e-5f

// ---------- bf16 helpers (integer-only) ----------
__device__ __forceinline__ u16 f2bf(float x) {
  unsigned u = __builtin_bit_cast(unsigned, x);
  unsigned r = u + 0x7FFFu + ((u >> 16) & 1u);   // round-to-nearest-even
  return (u16)(r >> 16);
}
__device__ __forceinline__ float bf2f(u16 h) {
  unsigned u = ((unsigned)h) << 16;
  return __builtin_bit_cast(float, u);
}

// ---------- WMMA fragment assembly (CDNA5 ISA 7.12.2, 16-bit, wave32) ----------
// A 16x32: lane half=lane>>4, row M=lane&15.
//   elements 0..7  -> K = half*8 + 0..7        (16B contiguous)
//   elements 8..15 -> K = 16 + half*8 + 0..7   (16B contiguous)
// B 32x16: col N=lane&15; elements 0..7 -> K = half*8 + 0..7, 8..15 -> K = 16 + half*8 + 0..7 (as A)
struct alignas(32) Frag32 { uint4 lo, hi; };
__device__ __forceinline__ v16u frag_from(uint4 lo, uint4 hi) {
  Frag32 f{lo, hi};
  return __builtin_bit_cast(v16u, f);
}
__device__ __forceinline__ uint4 zero4() { uint4 z; z.x = z.y = z.z = z.w = 0u; return z; }

__device__ __forceinline__ v8f wmma_bf16(v16u a, v16u b, v8f c) {
  return __builtin_amdgcn_wmma_f32_16x16x32_bf16(
      false, __builtin_bit_cast(v16bf, a),
      false, __builtin_bit_cast(v16bf, b),
      (short)0, c, false, false);
}

__device__ __forceinline__ float sigmoidf_(float x) { return 1.0f / (1.0f + expf(-x)); }

// =====================================================================
// Stage 0: direct conv Cin=1 (0.6 GFLOP), BN+ReLU fused, out NHWC bf16
// =====================================================================
__global__ void conv0a_kernel(const float* __restrict__ x, const float* __restrict__ w,
                              const float* __restrict__ cb, const float* __restrict__ g,
                              const float* __restrict__ be, const float* __restrict__ m,
                              const float* __restrict__ v, u16* __restrict__ out) {
  int idx = blockIdx.x * 256 + threadIdx.x;     // 64*128*128*32 threads
  int co = idx & 31;
  int p  = idx >> 5;                            // pixel = ((n*128)+oy)*128+ox
  int ox = p & 127;
  int r  = p >> 7;
  int oy = r & 127;
  int n  = r >> 7;
  float s    = g[co] * rsqrtf(v[co] + EPSLN);
  float bias = (cb[co] - m[co]) * s + be[co];
  float acc = 0.0f;
#pragma unroll
  for (int dy = -1; dy <= 1; ++dy) {
#pragma unroll
    for (int dx = -1; dx <= 1; ++dx) {
      int iy = oy + dy, ix = ox + dx;
      if (iy >= 0 && iy < 128 && ix >= 0 && ix < 128)
        acc += x[(size_t)(n * 128 + iy) * 128 + ix] * w[co * 9 + (dy + 1) * 3 + (dx + 1)];
    }
  }
  float y = fmaxf(acc * s + bias, 0.0f);
  out[idx] = f2bf(y);
}

// =====================================================================
// Weight pack: fold BN scale into bf16 weights; layout [tap][cout][cin]
// =====================================================================
__global__ void pack_conv_w_kernel(const float* __restrict__ w, const float* __restrict__ b,
                                   const float* __restrict__ g, const float* __restrict__ be,
                                   const float* __restrict__ m, const float* __restrict__ v,
                                   u16* __restrict__ wp, float* __restrict__ ba,
                                   int Cout, int Cin) {
  int idx = blockIdx.x * 256 + threadIdx.x;
  int total = 9 * Cout * Cin;
  if (idx >= total) return;
  int tap = idx / (Cout * Cin);
  int rem = idx - tap * Cout * Cin;
  int co  = rem / Cin;
  int ci  = rem - co * Cin;
  float s = g[co] * rsqrtf(v[co] + EPSLN);
  wp[idx] = f2bf(w[(size_t)(co * Cin + ci) * 9 + tap] * s);
  if (tap == 0 && ci == 0) ba[co] = (b[co] - m[co]) * s + be[co];
}

// =====================================================================
// Implicit-GEMM conv via WMMA bf16, 2x2 register blocking per wave:
//   wave computes 32 couts x 32 pixels (4 accumulators); fragments a0,a1,b0,b1
//   feed 4 WMMAs -> 2 memory ops per WMMA instead of 4.
// NHWC in/out. Cin%32==0, Cout%32==0. Folded weights (32 couts x 9 taps) staged
// in dynamic LDS (up to 73.7 KB of the 320 KB/WGP).
// =====================================================================
__global__ void __launch_bounds__(128)
conv_wmma_kernel(const u16* __restrict__ in, u16* __restrict__ out,
                 const u16* __restrict__ wpack, const float* __restrict__ biasAct,
                 int H, int W, int Cin, int Cout) {
  extern __shared__ u16 sW[];                    // 9*32*Cin u16
  const int tid  = threadIdx.x;
  const int lane = tid & 31, wave = tid >> 5;
  const int half = lane >> 4, r16 = lane & 15;
  const int coutBase = blockIdx.y * 32;

  // cooperative staging (uint4 = 8 u16 per transfer)
  const int total8 = (9 * 32 * Cin) >> 3;
  for (int i8 = tid; i8 < total8; i8 += 128) {
    int i   = i8 << 3;
    int tap = i / (32 * Cin);
    int rem = i - tap * 32 * Cin;
    int co  = rem / Cin;
    int ci  = rem - co * Cin;
    *(uint4*)(&sW[i]) =
        *(const uint4*)(&wpack[((size_t)tap * Cout + coutBase + co) * Cin + ci]);
  }
  __syncthreads();

  // two 16-pixel tiles per wave (32 consecutive pixels; 32 | 64*H*W per image)
  const int HW = H * W;
  const int pbase = (blockIdx.x * 4 + wave) * 32;
  int n0, oy0, oxB0, n1, oy1, oxB1;
  {
    int pb0 = pbase, pb1 = pbase + 16;
    n0 = pb0 / HW; int r0 = pb0 - n0 * HW; oy0 = r0 / W; oxB0 = r0 - oy0 * W;
    n1 = pb1 / HW; int r1 = pb1 - n1 * HW; oy1 = r1 / W; oxB1 = r1 - oy1 * W;
  }
  const int ox0 = oxB0 + r16, ox1 = oxB1 + r16;

  v8f acc00, acc01, acc10, acc11;
#pragma unroll
  for (int i = 0; i < 8; ++i) { acc00[i] = 0.0f; acc01[i] = 0.0f; acc10[i] = 0.0f; acc11[i] = 0.0f; }

  const int chunks = Cin >> 5;
  const int aOff = half * 8;                     // A: 16B run offsets within 32-chunk
  const int bOff = half * 8;                     // B: 16B run offsets within 32-chunk
  for (int tap = 0; tap < 9; ++tap) {
    int dy = tap / 3 - 1, dx = tap - (tap / 3) * 3 - 1;
    int iy0 = oy0 + dy, ix0 = ox0 + dx;
    int iy1 = oy1 + dy, ix1 = ox1 + dx;
    bool v0 = (iy0 >= 0) & (iy0 < H) & (ix0 >= 0) & (ix0 < W);
    bool v1 = (iy1 >= 0) & (iy1 < H) & (ix1 >= 0) & (ix1 < W);
    const u16* src0 = in + ((size_t)(n0 * H + iy0) * W + ix0) * Cin;  // NHWC
    const u16* src1 = in + ((size_t)(n1 * H + iy1) * W + ix1) * Cin;
    const u16* wrow0 = sW + (tap * 32 + r16) * Cin;                   // cout tile 0
    const u16* wrow1 = wrow0 + 16 * Cin;                              // cout tile 1
#pragma unroll 4
    for (int c = 0; c < chunks; ++c) {
      int cb = c << 5;
      uint4 a0lo = *(const uint4*)(wrow0 + cb + aOff);        // ds_load_b128 x4
      uint4 a0hi = *(const uint4*)(wrow0 + cb + 16 + aOff);
      uint4 a1lo = *(const uint4*)(wrow1 + cb + aOff);
      uint4 a1hi = *(const uint4*)(wrow1 + cb + 16 + aOff);
      uint4 b0lo = zero4(), b0hi = zero4(), b1lo = zero4(), b1hi = zero4();
      if (v0) {
        b0lo = *(const uint4*)(src0 + cb + bOff);             // global_load_b128
        b0hi = *(const uint4*)(src0 + cb + bOff + 16);
      }
      if (v1) {
        b1lo = *(const uint4*)(src1 + cb + bOff);
        b1hi = *(const uint4*)(src1 + cb + bOff + 16);
      }
      v16u a0 = frag_from(a0lo, a0hi), a1 = frag_from(a1lo, a1hi);
      v16u b0 = frag_from(b0lo, b0hi), b1 = frag_from(b1lo, b1hi);
      acc00 = wmma_bf16(a0, b0, acc00);
      acc01 = wmma_bf16(a0, b1, acc01);
      acc10 = wmma_bf16(a1, b0, acc10);
      acc11 = wmma_bf16(a1, b1, acc11);
    }
  }

  // Epilogue: D lane col = pixel (r16); VGPR r -> cout r + 8*half. Bias + ReLU.
  size_t op0 = ((size_t)(n0 * H + oy0) * W + (oxB0 + r16)) * Cout + coutBase;
  size_t op1 = ((size_t)(n1 * H + oy1) * W + (oxB1 + r16)) * Cout + coutBase;
#pragma unroll
  for (int r = 0; r < 8; ++r) {
    int c0 = r + 8 * half;                        // cout tile 0 channel
    float ba0 = biasAct[coutBase + c0];
    float ba1 = biasAct[coutBase + 16 + c0];
    out[op0 + c0]      = f2bf(fmaxf(acc00[r] + ba0, 0.0f));
    out[op1 + c0]      = f2bf(fmaxf(acc01[r] + ba0, 0.0f));
    out[op0 + 16 + c0] = f2bf(fmaxf(acc10[r] + ba1, 0.0f));
    out[op1 + 16 + c0] = f2bf(fmaxf(acc11[r] + ba1, 0.0f));
  }
}

// =====================================================================
// 2x2 max pool, NHWC bf16
// =====================================================================
__global__ void pool_kernel(const u16* __restrict__ in, u16* __restrict__ out,
                            int H, int W, int C, int total) {
  int idx = blockIdx.x * 256 + threadIdx.x;
  if (idx >= total) return;
  int c = idx % C;
  int r = idx / C;
  int Wo = W >> 1, Ho = H >> 1;
  int x = r % Wo; r /= Wo;
  int y = r % Ho;
  int n = r / Ho;
  size_t b00 = ((size_t)(n * H + 2 * y) * W + 2 * x) * C + c;
  float m0 = fmaxf(bf2f(in[b00]), bf2f(in[b00 + C]));
  float m1 = fmaxf(bf2f(in[b00 + (size_t)W * C]), bf2f(in[b00 + (size_t)W * C + C]));
  out[idx] = f2bf(fmaxf(m0, m1));
}

// =====================================================================
// Projection LN: gather seq[b,t,d] = f_nhwc[b, d%16, t, d/16], LN over 2048,
// write bf16 rows [b*16+t][2048]
// =====================================================================
__global__ void ln_proj_kernel(const u16* __restrict__ f, const float* __restrict__ g,
                               const float* __restrict__ b, u16* __restrict__ out) {
  int row = blockIdx.x;              // 1024 rows = b*16 + t
  int bb = row >> 4, t = row & 15;
  int tid = threadIdx.x;             // 256 threads
  float vals[8], s = 0.0f, sq = 0.0f;
#pragma unroll
  for (int j = 0; j < 8; ++j) {
    int d = tid + j * 256;
    float x = bf2f(f[((size_t)(bb * 16 + (d & 15)) * 16 + t) * 128 + (d >> 4)]);
    vals[j] = x; s += x; sq += x * x;
  }
  for (int msk = 16; msk >= 1; msk >>= 1) {
    s  += __shfl_xor(s,  msk, 32);
    sq += __shfl_xor(sq, msk, 32);
  }
  __shared__ float ps[8], pq[8];
  int lane = tid & 31, wv = tid >> 5;
  if (lane == 0) { ps[wv] = s; pq[wv] = sq; }
  __syncthreads();
  s = 0.0f; sq = 0.0f;
#pragma unroll
  for (int w = 0; w < 8; ++w) { s += ps[w]; sq += pq[w]; }
  float mean = s * (1.0f / 2048.0f);
  float var  = sq * (1.0f / 2048.0f) - mean * mean;
  float rstd = rsqrtf(var + EPSLN);
#pragma unroll
  for (int j = 0; j < 8; ++j) {
    int d = tid + j * 256;
    out[(size_t)row * 2048 + d] = f2bf((vals[j] - mean) * rstd * g[d] + b[d]);
  }
}

// =====================================================================
// LN over 256 cols (one wave per row), f32 in -> bf16 out
// =====================================================================
__global__ void ln256_kernel(const float* __restrict__ in, const float* __restrict__ g,
                             const float* __restrict__ b, u16* __restrict__ out) {
  int row = blockIdx.x;
  int lane = threadIdx.x;            // 32 threads
  float v[8], s = 0.0f, sq = 0.0f;
#pragma unroll
  for (int j = 0; j < 8; ++j) {
    float x = in[(size_t)row * 256 + lane + j * 32];
    v[j] = x; s += x; sq += x * x;
  }
  for (int msk = 16; msk >= 1; msk >>= 1) {
    s  += __shfl_xor(s,  msk, 32);
    sq += __shfl_xor(sq, msk, 32);
  }
  float mean = s * (1.0f / 256.0f);
  float var  = sq * (1.0f / 256.0f) - mean * mean;
  float rstd = rsqrtf(var + EPSLN);
#pragma unroll
  for (int j = 0; j < 8; ++j) {
    int c = lane + j * 32;
    out[(size_t)row * 256 + c] = f2bf((v[j] - mean) * rstd * g[c] + b[c]);
  }
}

// =====================================================================
// Dual GEMM (WMMA bf16): C[m,n] = A1·W1^T + A2·W2^T + bias1 (+bias2)
// blockIdx.y -> 16-row M tile (A staged in LDS, K tiled by 512);
// blockIdx.x*4+wave -> 16-col N tile. Requires (K1+K2)%512==0, K1%32==0.
// mode: 0=raw, 2=exact GELU. scatter_bt: row b*16+t -> t*64+b.
// =====================================================================
__global__ void __launch_bounds__(128)
dual_gemm_kernel(const u16* __restrict__ A1, const u16* __restrict__ A2,
                 const u16* __restrict__ W1, int ldw1,
                 const u16* __restrict__ W2, int ldw2,
                 const float* __restrict__ bias1, const float* __restrict__ bias2,
                 float* __restrict__ out, int ldo,
                 int K1, int K2, int mode, int scatter_bt) {
  __shared__ alignas(16) u16 sA[16 * 512];       // 16 KB
  const int K    = K1 + K2;
  const int tid  = threadIdx.x;
  const int lane = tid & 31, wave = tid >> 5;
  const int half = lane >> 4, r16 = lane & 15;
  const int mBase = blockIdx.y * 16;
  const int nBase = (blockIdx.x * 4 + wave) * 16;
  const int nCol  = nBase + r16;
  const int aOff = half * 8;
  const int bOff = half * 8;

  __builtin_prefetch(W1 + (size_t)nCol * ldw1, 0, 0);

  v8f acc;
#pragma unroll
  for (int i = 0; i < 8; ++i) acc[i] = 0.0f;

  for (int kt = 0; kt < K; kt += 512) {
    __syncthreads();
    for (int i8 = tid; i8 < (16 * 512) >> 3; i8 += 128) {
      int i  = i8 << 3;
      int rr = i >> 9, kk = i & 511;
      int gk = kt + kk;                          // K1%8==0 -> whole uint4 on one side
      const u16* s = (gk < K1) ? &A1[(size_t)(mBase + rr) * K1 + gk]
                               : &A2[(size_t)(mBase + rr) * K2 + (gk - K1)];
      *(uint4*)(&sA[i]) = *(const uint4*)s;
    }
    __syncthreads();
#pragma unroll 4
    for (int c = 0; c < 16; ++c) {
      int kbL = c * 32;
      int kbG = kt + kbL;                        // chunk fully inside K1 or K2 (K1%32==0)
      const u16* wsrc = (kbG < K1) ? (W1 + (size_t)nCol * ldw1 + kbG)
                                   : (W2 + (size_t)nCol * ldw2 + (kbG - K1));
      const u16* arow = &sA[r16 * 512 + kbL];
      uint4 alo = *(const uint4*)(arow + aOff);            // ds_load_b128
      uint4 ahi = *(const uint4*)(arow + 16 + aOff);
      uint4 blo = *(const uint4*)(wsrc + bOff);            // global_load_b128
      uint4 bhi = *(const uint4*)(wsrc + bOff + 16);
      acc = wmma_bf16(frag_from(alo, ahi), frag_from(blo, bhi), acc);
    }
  }

  float b1v = bias1 ? bias1[nCol] : 0.0f;
  float b2v = bias2 ? bias2[nCol] : 0.0f;
#pragma unroll
  for (int r = 0; r < 8; ++r) {
    int mm = mBase + r + 8 * half;
    float x = acc[r] + b1v + b2v;
    if (mode == 2) x = 0.5f * x * (1.0f + erff(x * 0.70710678118654752f));   // exact GELU
    int rowOut = scatter_bt ? ((mm & 15) * 64 + (mm >> 4)) : mm;
    out[(size_t)rowOut * ldo + nCol] = x;
  }
}

// =====================================================================
// LSTM pointwise: gates -> (h,c) update with r/m residual gates
// =====================================================================
__global__ void lstm_point_kernel(const float* __restrict__ gates, const float* __restrict__ rgo,
                                  const float* __restrict__ mgo,
                                  float* __restrict__ hs, float* __restrict__ cs) {
  int i = blockIdx.x * 256 + threadIdx.x;        // 64*256
  int b = i >> 8, j = i & 255;
  const float* gr = gates + (size_t)b * 1024;
  float ii = sigmoidf_(gr[j]);
  float ff = sigmoidf_(gr[256 + j]);
  float gg = tanhf(gr[512 + j]);
  float oo = sigmoidf_(gr[768 + j]);
  float c_old = cs[i], h_old = hs[i];
  float c_new = ff * c_old + ii * gg;
  float tc    = tanhf(c_new);
  float h_new = oo * tc;
  float rg = sigmoidf_(rgo[i]);
  float h_out = rg * h_new + (1.0f - rg) * h_old;
  float mg = sigmoidf_(mgo[i]);
  float c_out = mg * c_new + (1.0f - mg) * tc;
  hs[i] = h_out;
  cs[i] = c_out;
}

// =====================================================================
// Head: LN(256) then 256->5 linear, one wave per batch row
// =====================================================================
__global__ void head_kernel(const float* __restrict__ h, const float* __restrict__ g,
                            const float* __restrict__ b, const float* __restrict__ w,
                            const float* __restrict__ bias, float* __restrict__ out) {
  int row = blockIdx.x;
  int lane = threadIdx.x;
  float v[8], s = 0.0f, sq = 0.0f;
#pragma unroll
  for (int j = 0; j < 8; ++j) {
    float x = h[(size_t)row * 256 + lane + j * 32];
    v[j] = x; s += x; sq += x * x;
  }
  for (int msk = 16; msk >= 1; msk >>= 1) {
    s  += __shfl_xor(s,  msk, 32);
    sq += __shfl_xor(sq, msk, 32);
  }
  float mean = s * (1.0f / 256.0f);
  float var  = sq * (1.0f / 256.0f) - mean * mean;
  float rstd = rsqrtf(var + EPSLN);
  float xn[8];
#pragma unroll
  for (int j = 0; j < 8; ++j) {
    int c = lane + j * 32;
    xn[j] = (v[j] - mean) * rstd * g[c] + b[c];
  }
  for (int cls = 0; cls < 5; ++cls) {
    float p = 0.0f;
#pragma unroll
    for (int j = 0; j < 8; ++j) p += xn[j] * w[cls * 256 + lane + j * 32];
    for (int msk = 16; msk >= 1; msk >>= 1) p += __shfl_xor(p, msk, 32);
    if (lane == 0) out[row * 5 + cls] = p + bias[cls];
  }
}

// ---------- misc small kernels ----------
__global__ void pack_bf_kernel(const float* __restrict__ in, u16* __restrict__ out, int n) {
  int i = blockIdx.x * 256 + threadIdx.x;
  if (i < n) out[i] = f2bf(in[i]);
}
__global__ void fill_zero_kernel(float* __restrict__ p, int n) {
  int i = blockIdx.x * 256 + threadIdx.x;
  if (i < n) p[i] = 0.0f;
}

// =====================================================================
// Launcher
// =====================================================================
extern "C" void kernel_launch(void* const* d_in, const int* in_sizes, int n_in,
                              void* d_out, int out_size, void* d_ws, size_t ws_size,
                              hipStream_t stream) {
  (void)in_sizes; (void)n_in; (void)out_size; (void)ws_size;
  auto F = [&](int i) { return (const float*)d_in[i]; };
  const float* X = F(0);

  // --- workspace carve ---
  char* p = (char*)d_ws;
  auto carve = [&](size_t bytes) -> char* {
    char* r = p;
    p += (bytes + 255) & ~(size_t)255;
    return r;
  };
  u16*   bufA    = (u16*)  carve((size_t)33554432 * 2);   // 64*128*128*32 bf16
  u16*   bufB    = (u16*)  carve((size_t)33554432 * 2);
  u16*   wpack   = (u16*)  carve((size_t)147456 * 2);     // 9*128*128 bf16
  float* biasAct = (float*)carve(128 * 4);
  u16*   xnProj  = (u16*)  carve((size_t)1024 * 2048 * 2);
  u16*   wProjBf = (u16*)  carve((size_t)524288 * 2);
  float* seq     = (float*)carve((size_t)16 * 64 * 256 * 4);
  u16*   wIhBf[2], *wHhBf[2], *rgWBf[2], *mgWBf[2];
  for (int L = 0; L < 2; ++L) {
    wIhBf[L] = (u16*)carve((size_t)262144 * 2);
    wHhBf[L] = (u16*)carve((size_t)262144 * 2);
    rgWBf[L] = (u16*)carve((size_t)131072 * 2);
    mgWBf[L] = (u16*)carve((size_t)131072 * 2);
  }
  float* gates = (float*)carve((size_t)64 * 1024 * 4);
  float* rgo   = (float*)carve((size_t)64 * 256 * 4);
  float* mgo   = (float*)carve((size_t)64 * 256 * 4);
  u16*   xn    = (u16*)  carve((size_t)64 * 256 * 2);
  u16*   hn    = (u16*)  carve((size_t)64 * 256 * 2);
  float* state = (float*)carve((size_t)4 * 64 * 256 * 4); // hs0,cs0,hs1,cs1
  float* hs0 = state, *cs0 = state + 16384, *hs1 = state + 32768, *cs1 = state + 49152;

  // --- conv helper (2x2 blocked conv: 32 couts x 32 pixels per wave) ---
  auto runConv = [&](const u16* in, u16* out, int H, int W, int Cin, int Cout, int wIdx) {
    int tot = 9 * Cout * Cin;
    pack_conv_w_kernel<<<(tot + 255) / 256, 256, 0, stream>>>(
        F(wIdx), F(wIdx + 1), F(wIdx + 2), F(wIdx + 3), F(wIdx + 4), F(wIdx + 5),
        wpack, biasAct, Cout, Cin);
    int waveTiles = (64 * H * W) / 32;            // 32 pixels per wave
    dim3 grid(waveTiles / 4, Cout / 32);
    size_t shBytes = (size_t)9 * 32 * Cin * 2;    // dynamic LDS: folded weights
    conv_wmma_kernel<<<grid, 128, shBytes, stream>>>(in, out, wpack, biasAct, H, W, Cin, Cout);
  };
  auto runPool = [&](const u16* in, u16* out, int H, int W, int C) {
    int tot = 64 * (H / 2) * (W / 2) * C;
    pool_kernel<<<(tot + 255) / 256, 256, 0, stream>>>(in, out, H, W, C, tot);
  };

  // --- backbone (NHWC bf16 activations, ping-pong in L2-resident buffers) ---
  conv0a_kernel<<<131072, 256, 0, stream>>>(X, F(1), F(2), F(3), F(4), F(5), F(6), bufA);
  runConv(bufA, bufB, 128, 128, 32, 32, 7);      // block0 conv2
  runPool(bufB, bufA, 128, 128, 32);             // -> 64x64x32
  runConv(bufA, bufB, 64, 64, 32, 64, 13);       // block1 conv1
  runConv(bufB, bufA, 64, 64, 64, 64, 19);       // block1 conv2
  runPool(bufA, bufB, 64, 64, 64);               // -> 32x32x64
  runConv(bufB, bufA, 32, 32, 64, 128, 25);      // block2 conv1
  runConv(bufA, bufB, 32, 32, 128, 128, 31);     // block2 conv2
  runPool(bufB, bufA, 32, 32, 128);              // -> 16x16x128
  runConv(bufA, bufB, 16, 16, 128, 128, 37);     // block3 conv1
  runConv(bufB, bufA, 16, 16, 128, 128, 43);     // block3 conv2 -> f in bufA

  // --- projection: LN(2048) -> GEMM(2048->256) + GELU, scatter to [t][b][256] ---
  ln_proj_kernel<<<1024, 256, 0, stream>>>(bufA, F(49), F(50), xnProj);
  pack_bf_kernel<<<524288 / 256, 256, 0, stream>>>(F(51), wProjBf, 524288);
  dual_gemm_kernel<<<dim3(256 / 64, 1024 / 16), 128, 0, stream>>>(
      xnProj, nullptr, wProjBf, 2048, nullptr, 0, F(52), nullptr,
      seq, 256, 2048, 0, /*mode=*/2, /*scatter_bt=*/1);

  // --- pack LSTM weights to bf16 ---
  for (int L = 0; L < 2; ++L) {
    int base = 53 + 12 * L;
    pack_bf_kernel<<<262144 / 256, 256, 0, stream>>>(F(base + 4),  wIhBf[L], 262144);
    pack_bf_kernel<<<262144 / 256, 256, 0, stream>>>(F(base + 6),  wHhBf[L], 262144);
    pack_bf_kernel<<<131072 / 256, 256, 0, stream>>>(F(base + 8),  rgWBf[L], 131072);
    pack_bf_kernel<<<131072 / 256, 256, 0, stream>>>(F(base + 10), mgWBf[L], 131072);
  }
  fill_zero_kernel<<<(4 * 16384) / 256, 256, 0, stream>>>(state, 4 * 16384);

  // --- LSTM: T=16, 2 layers, sequential on stream ---
  for (int t = 0; t < 16; ++t) {
    for (int L = 0; L < 2; ++L) {
      int base = 53 + 12 * L;
      const float* inp = (L == 0) ? (seq + (size_t)t * 64 * 256) : hs0;
      float* hsL = (L == 0) ? hs0 : hs1;
      float* csL = (L == 0) ? cs0 : cs1;
      ln256_kernel<<<64, 32, 0, stream>>>(inp, F(base + 0), F(base + 1), xn);
      ln256_kernel<<<64, 32, 0, stream>>>(hsL, F(base + 2), F(base + 3), hn);
      // gates = xn@w_ih^T + hn@w_hh^T + b_ih + b_hh   (M=64, N=1024, K=256+256)
      dual_gemm_kernel<<<dim3(16, 4), 128, 0, stream>>>(
          xn, hn, wIhBf[L], 256, wHhBf[L], 256, F(base + 5), F(base + 7),
          gates, 1024, 256, 256, 0, 0);
      // r = xh@rg_w^T + rg_b ; xh=[xn|hn], rg_w is [256,512]
      dual_gemm_kernel<<<dim3(4, 4), 128, 0, stream>>>(
          xn, hn, rgWBf[L], 512, rgWBf[L] + 256, 512, F(base + 9), nullptr,
          rgo, 256, 256, 256, 0, 0);
      dual_gemm_kernel<<<dim3(4, 4), 128, 0, stream>>>(
          xn, hn, mgWBf[L], 512, mgWBf[L] + 256, 512, F(base + 11), nullptr,
          mgo, 256, 256, 256, 0, 0);
      lstm_point_kernel<<<64, 256, 0, stream>>>(gates, rgo, mgo, hsL, csL);
    }
  }

  // --- head: LN + 256->5 ---
  head_kernel<<<64, 32, 0, stream>>>(hs1, F(77), F(78), F(79), F(80), (float*)d_out);
}
